// conv3d_69123203662242
// MI455X (gfx1250) — matrix operand order repaired and checked by reading, not compiled, not run on hardware
//
#include <hip/hip_runtime.h>

// Submanifold sparse conv3d for MI455X (gfx1250, wave32, WMMA).
// out[i] = sum_k feats[nbr_k(i)] @ kernel[k]  (27 taps, Cin=Cout=64)
//
// Strategy:
//  - open-addressing hash (1M slots, 8 MB -> L2 resident) for voxel -> index
//  - feats pre-split ONCE into f16 hi/lo arrays (fp32-class accuracy via
//      A*B ~= Ah*Bh + Al*Bh + Ah*Bl, error ~2^-22); removes all cvt VALU
//      from the hot loop
//  - invalid neighbors gather from a dedicated zero row (index N) -> no
//      register masking in the hot loop at all
//  - one wave32 per 16-point tile; 16x64 fp32 accumulator (4 WMMA C-tiles)
//  - weights pre-swizzled into exact B-fragment lane layout (hi & lo copies)
//  - per-tile tap skipping via wave ballot (occupancy ~1.6% -> ~7/27 taps run)

#define HLOG  20
#define HSIZE (1u << HLOG)
#define GRIDE 256

typedef __attribute__((ext_vector_type(16))) _Float16 v16h;
typedef __attribute__((ext_vector_type(8)))  _Float16 v8h;
typedef __attribute__((ext_vector_type(4)))  _Float16 v4h;
typedef __attribute__((ext_vector_type(8)))  float    v8f;

__global__ void hash_init_kernel(int* __restrict__ keys, int* __restrict__ vals) {
    unsigned i = blockIdx.x * blockDim.x + threadIdx.x;
    if (i < HSIZE) { keys[i] = -1; vals[i] = -1; }
}

__global__ void hash_insert_kernel(const int* __restrict__ coords,
                                   int* __restrict__ keys, int* __restrict__ vals,
                                   int N) {
    int i = blockIdx.x * blockDim.x + threadIdx.x;
    if (i >= N) return;
    unsigned lin = (((unsigned)coords[3*i] * GRIDE) + (unsigned)coords[3*i+1]) * GRIDE
                   + (unsigned)coords[3*i+2];
    unsigned h = (lin * 2654435761u) >> (32 - HLOG);
    for (unsigned p = 0; p < HSIZE; ++p) {
        int old = atomicCAS(&keys[h], -1, (int)lin);
        if (old == -1 || old == (int)lin) {
            atomicMax(&vals[h], i);   // deterministic winner for duplicate voxels
            return;
        }
        h = (h + 1) & (HSIZE - 1);
    }
}

// Split feats (fp32 [N][64]) into f16 hi + f16 lo, row-major, 4 elems/thread.
// Row N (one past the end) is written as all-zeros: gather target for
// invalid/out-of-range neighbors.
__global__ void feat_split_kernel(const float* __restrict__ feats,
                                  _Float16* __restrict__ fh,
                                  _Float16* __restrict__ fl,
                                  int total4, int zeros4) {
    int i = blockIdx.x * blockDim.x + threadIdx.x;
    if (i >= total4 + zeros4) return;
    v4h hi = {0, 0, 0, 0}, lo = {0, 0, 0, 0};
    if (i < total4) {
        float4 x = *(const float4*)(feats + (size_t)i * 4);
        float xs[4] = {x.x, x.y, x.z, x.w};
#pragma unroll
        for (int t = 0; t < 4; ++t) {
            _Float16 h = (_Float16)xs[t];
            hi[t] = h;
            lo[t] = (_Float16)(xs[t] - (float)h);
        }
    }
    *(v4h*)(fh + (size_t)i * 4) = hi;
    *(v4h*)(fl + (size_t)i * 4) = lo;
}

// Pre-swizzle kernel weights (fp32 [27][64][64]) into WMMA B-fragment layout,
// split into f16 hi + f16 lo. B 32x16 f16 layout: lane<16 holds column N=lane,
// K={0..7,16..23} (t=0..15); lane>=16 holds column N=lane-16, K={8..15,24..31}.
__global__ void weight_swizzle_kernel(const float* __restrict__ kern,
                                      _Float16* __restrict__ wh,
                                      _Float16* __restrict__ wl) {
    int e = blockIdx.x * blockDim.x + threadIdx.x;
    if (e >= 27 * 64 * 64) return;
    int t    =  e        & 15;
    int lane = (e >> 4)  & 31;
    int nb   = (e >> 9)  & 3;
    int kb   = (e >> 11) & 1;
    int k    =  e >> 12;
    int kk = ((t < 8) ? t : (t + 8)) + ((lane < 16) ? 0 : 8);
    int ci = kb * 32 + kk;
    int co = nb * 16 + (lane & 15);
    float w = kern[(k * 64 + ci) * 64 + co];
    _Float16 hi = (_Float16)w;
    _Float16 lo = (_Float16)(w - (float)hi);
    wh[e] = hi;
    wl[e] = lo;
}

__global__ void __launch_bounds__(256)
sparse_conv_kernel(const int* __restrict__ coords,
                   const _Float16* __restrict__ fh,
                   const _Float16* __restrict__ fl,
                   const int* __restrict__ keys,
                   const int* __restrict__ vals,
                   const _Float16* __restrict__ wh,
                   const _Float16* __restrict__ wl,
                   float* __restrict__ out, int N) {
    int wave = blockIdx.x * (blockDim.x >> 5) + (threadIdx.x >> 5);
    int lane = threadIdx.x & 31;
    int ntiles = (N + 15) >> 4;
    if (wave >= ntiles) return;               // wave-uniform

    int m   = lane & 15;                      // tile row this lane serves
    int row = wave * 16 + m;
    int rc  = row < N ? row : N - 1;
    bool rowok = row < N;
    int cx = coords[3*rc], cy = coords[3*rc+1], cz = coords[3*rc+2];

    v8f zero = {0.f,0.f,0.f,0.f,0.f,0.f,0.f,0.f};
    v8f acc[4];
#pragma unroll
    for (int nb = 0; nb < 4; ++nb) acc[nb] = zero;

#pragma unroll 1
    for (int k = 0; k < 27; ++k) {
        int dx = k / 9 - 1, dy = (k / 3) % 3 - 1, dz = k % 3 - 1;
        int nx = cx + dx, ny = cy + dy, nz = cz + dz;
        bool inb = rowok && ((unsigned)nx < GRIDE) && ((unsigned)ny < GRIDE)
                         && ((unsigned)nz < GRIDE);
        int nbr = -1;
        if (inb) {
            unsigned lin = (((unsigned)nx * GRIDE) + (unsigned)ny) * GRIDE + (unsigned)nz;
            unsigned h = (lin * 2654435761u) >> (32 - HLOG);
            while (true) {
                int key = keys[h];
                if (key == (int)lin) { nbr = vals[h]; break; }
                if (key == -1) break;
                h = (h + 1) & (HSIZE - 1);
            }
        }
        bool valid = nbr >= 0;
        if (__ballot(valid) == 0ull) continue;   // skip empty tap (wave-uniform)

        // invalid lanes gather from the all-zero row at index N
        size_t roff = (size_t)(valid ? nbr : N) * 64;
        const _Float16* fh_row = fh + roff;
        const _Float16* fl_row = fl + roff;

#pragma unroll
        for (int kb = 0; kb < 2; ++kb) {
            // A 16x32 f16 layout: lane<16 -> K {0..7,16..23}; lane>=16 -> {8..15,24..31}
            int base = kb * 32 + ((lane < 16) ? 0 : 8);
            v8h ah0 = *(const v8h*)(fh_row + base);
            v8h ah1 = *(const v8h*)(fh_row + base + 16);
            v8h al0 = *(const v8h*)(fl_row + base);
            v8h al1 = *(const v8h*)(fl_row + base + 16);
            v16h ah = __builtin_shufflevector(ah0, ah1, 0,1,2,3,4,5,6,7,
                                              8,9,10,11,12,13,14,15);
            v16h al = __builtin_shufflevector(al0, al1, 0,1,2,3,4,5,6,7,
                                              8,9,10,11,12,13,14,15);

            const _Float16* bh_base = wh + ((size_t)((k * 2 + kb) * 4) * 32 + lane) * 16;
            const _Float16* bl_base = wl + ((size_t)((k * 2 + kb) * 4) * 32 + lane) * 16;
#pragma unroll
            for (int nb = 0; nb < 4; ++nb) {
                v16h bh = *(const v16h*)(bh_base + (size_t)nb * 32 * 16);
                v16h bl = *(const v16h*)(bl_base + (size_t)nb * 32 * 16);
                acc[nb] = __builtin_amdgcn_wmma_f32_16x16x32_f16(
                    false, ah, false, bh, (short)0, acc[nb], false, false);
                acc[nb] = __builtin_amdgcn_wmma_f32_16x16x32_f16(
                    false, al, false, bh, (short)0, acc[nb], false, false);
                acc[nb] = __builtin_amdgcn_wmma_f32_16x16x32_f16(
                    false, ah, false, bl, (short)0, acc[nb], false, false);
            }
        }
    }

    // C/D 16x16 f32 layout: VGPR r -> M=r (lane<16) / M=r+8 (lane>=16), N=lane&15
    int mrow0 = wave * 16 + ((lane < 16) ? 0 : 8);
    int co0 = lane & 15;
#pragma unroll
    for (int nb = 0; nb < 4; ++nb) {
#pragma unroll
        for (int r = 0; r < 8; ++r) {
            int orow = mrow0 + r;
            if (orow < N) out[(size_t)orow * 64 + nb * 16 + co0] = acc[nb][r];
        }
    }
}

extern "C" void kernel_launch(void* const* d_in, const int* in_sizes, int n_in,
                              void* d_out, int out_size, void* d_ws, size_t ws_size,
                              hipStream_t stream) {
    const int*   coords = (const int*)d_in[0];
    const float* feats  = (const float*)d_in[1];
    const float* kern   = (const float*)d_in[2];
    float* out = (float*)d_out;
    int N = in_sizes[0] / 3;

    char* ws = (char*)d_ws;
    int* keys = (int*)ws;                                    // [0, 4 MB)
    int* vals = (int*)(ws + (size_t)HSIZE * 4);              // [4, 8 MB)
    _Float16* wh = (_Float16*)(ws + ((size_t)8 << 20));                  // 221184 B
    _Float16* wl = (_Float16*)(ws + ((size_t)8 << 20) + (256u << 10));   // 221184 B
    _Float16* fh = (_Float16*)(ws + ((size_t)16 << 20));     // (N+1)*64*2 B
    _Float16* fl = (_Float16*)(ws + ((size_t)48 << 20));     // (N+1)*64*2 B

    hash_init_kernel<<<(HSIZE + 255) / 256, 256, 0, stream>>>(keys, vals);
    hash_insert_kernel<<<(N + 255) / 256, 256, 0, stream>>>(coords, keys, vals, N);
    weight_swizzle_kernel<<<(27 * 64 * 64 + 255) / 256, 256, 0, stream>>>(kern, wh, wl);
    int total4 = N * 64 / 4;        // real feature elements / 4
    int zeros4 = 64 / 4;            // one extra all-zero row at index N
    feat_split_kernel<<<(total4 + zeros4 + 255) / 256, 256, 0, stream>>>(
        feats, fh, fl, total4, zeros4);

    int ntiles = (N + 15) / 16;
    int waves_per_block = 8;                  // 256 threads = 8 wave32
    int blocks = (ntiles + waves_per_block - 1) / waves_per_block;
    sparse_conv_kernel<<<blocks, 256, 0, stream>>>(coords, fh, fl, keys, vals,
                                                   wh, wl, out, N);
}
